// WAP_76544907149446
// MI455X (gfx1250) — compile-verified, matrix operand order checked by reading, not run on hardware
//
#include <hip/hip_runtime.h>
#include <hip/hip_bf16.h>

// ---------------------------------------------------------------------------
// CDNA5 WMMA types (wave32): A/B = 16 bf16 (8 VGPRs), C/D = 8 f32
// ---------------------------------------------------------------------------
typedef __attribute__((ext_vector_type(16))) __bf16 v16bf;
typedef __attribute__((ext_vector_type(8)))  float  v8f;

union BF16x16 { v16bf v; unsigned short u[16]; uint4 q[2]; };

// f32 -> bf16 via HIP intrinsic (lowers to native v_cvt on gfx1250)
__device__ __forceinline__ unsigned short f2bf(float f) {
    __hip_bfloat16 h = __float2bfloat16(f);
    return __builtin_bit_cast(unsigned short, h);
}
__device__ __forceinline__ float bf2f(unsigned short b) {
    return __uint_as_float(((unsigned int)b) << 16);
}
__device__ __forceinline__ v8f zero8() {
    v8f v = {0.f,0.f,0.f,0.f,0.f,0.f,0.f,0.f};
    return v;
}

// ---------------------------------------------------------------------------
// Packing / utility kernels
// ---------------------------------------------------------------------------
__global__ void f32_to_bf16_kernel(const float* __restrict__ in,
                                   unsigned short* __restrict__ out, int n) {
    int i = blockIdx.x * 256 + threadIdx.x;
    if (i < n) out[i] = f2bf(in[i]);
}

__global__ void zero_u16_kernel(unsigned short* __restrict__ p, int n) {
    int i = blockIdx.x * 256 + threadIdx.x;
    if (i < n) p[i] = 0;
}

// conv weight (CO,CI,3,3) f32 -> (9, CO, CIP) bf16, channel-padded with zeros
__global__ void pack_conv_w(const float* __restrict__ w,
                            unsigned short* __restrict__ dst,
                            int CO, int CI, int CIP) {
    int i = blockIdx.x * 256 + threadIdx.x;
    int total = 9 * CO * CIP;
    if (i >= total) return;
    int ci = i % CIP; int tmp = i / CIP;
    int co = tmp % CO; int r9 = tmp / CO;
    unsigned short v = 0;
    if (ci < CI) v = f2bf(w[((size_t)co * CI + ci) * 9 + r9]);
    dst[i] = v;
}

// input (8,9,H,W) f32 NCHW -> (8,H,W,32) bf16 NHWC, channels 9..31 zero
__global__ void pack_input_nhwc(const float* __restrict__ x,
                                unsigned short* __restrict__ dst, int H, int W) {
    int i = blockIdx.x * 256 + threadIdx.x;
    int total = 8 * H * W * 32;
    if (i >= total) return;
    int c = i & 31; int pos = i >> 5;
    int xw = pos % W; int tmp = pos / W;
    int y = tmp % H;  int b = tmp / H;
    unsigned short v = 0;
    if (c < 9) v = f2bf(x[((size_t)(b * 9 + c) * H + y) * W + xw]);
    dst[i] = v;
}

// ---------------------------------------------------------------------------
// 3x3 SAME conv + bias + ReLU as 9 shifted GEMMs, NHWC bf16, WMMA f32 accum.
// H,W are template params so all tap/K offsets fold into immediate offsets.
// Border lanes redirect their A pointer to a zeroed scratch line (zbuf).
// ---------------------------------------------------------------------------
template<int CIP, int CO, int MT, int H, int W>
__global__ __launch_bounds__(32)
void conv3x3_nhwc_wmma(const unsigned short* __restrict__ in,  // (B,H,W,CIP)
                       const unsigned short* __restrict__ wp,  // (9,CO,CIP)
                       const float* __restrict__ bias,         // (CO)
                       unsigned short* __restrict__ out,       // (B,H,W,CO)
                       const unsigned short* __restrict__ zbuf) {
    const int lane = threadIdx.x & 31;
    const int m    = lane & 15;
    const int g    = lane >> 4;
    const int n0   = blockIdx.y * 16;
    constexpr int HW = H * W;

    v8f acc[MT];
#pragma unroll
    for (int t = 0; t < MT; ++t) acc[t] = zero8();

    int yy[MT];
    const unsigned short* abase[MT];
    unsigned short* ob[MT];
    bool vldL[MT], vldR[MT];
#pragma unroll
    for (int t = 0; t < MT; ++t) {
        int p  = (blockIdx.x * MT + t) * 16;
        int b  = p / HW;
        int rm = p - b * HW;
        yy[t]  = rm / W;
        int x0 = rm - yy[t] * W;
        int xt = x0 + m;
        abase[t] = in + ((size_t)(b * H + yy[t]) * W + xt) * CIP + g * 16;
        ob[t]    = out + ((size_t)(b * H + yy[t]) * W + x0 + g * 8) * CO + n0 + m;
        vldL[t]  = (xt - 1) >= 0;
        vldR[t]  = (xt + 1) < W;
    }

    const unsigned short* wbase = wp + (size_t)(n0 + m) * CIP + g * 16;

#pragma unroll
    for (int ky = 0; ky < 3; ++ky) {
#pragma unroll
        for (int kx = 0; kx < 3; ++kx) {
            const unsigned short* wtap =
                wbase + (size_t)((ky * 3 + kx) * CO) * CIP;
            const unsigned short* ap[MT];
            bool iyok[MT];
#pragma unroll
            for (int t = 0; t < MT; ++t) {
                int iy  = yy[t] + ky - 1;
                iyok[t] = (iy >= 0) && (iy < H);        // wave-uniform
                bool vld = (kx == 0) ? vldL[t] : ((kx == 2) ? vldR[t] : true);
                ap[t] = vld ? (abase[t] + ((ky - 1) * W + (kx - 1)) * CIP)
                            : zbuf;
            }
#pragma unroll
            for (int kb = 0; kb < CIP; kb += 32) {
                BF16x16 bf;
                bf.q[0] = *(const uint4*)(wtap + kb);
                bf.q[1] = *(const uint4*)(wtap + kb + 8);
#pragma unroll
                for (int t = 0; t < MT; ++t) {
                    if (!iyok[t]) continue;             // scalar skip, EXEC full
                    BF16x16 af;
                    af.q[0] = *(const uint4*)(ap[t] + kb);
                    af.q[1] = *(const uint4*)(ap[t] + kb + 8);
                    acc[t] = __builtin_amdgcn_wmma_f32_16x16x32_bf16(
                        false, af.v, false, bf.v, (short)0, acc[t], false, false);
                }
            }
        }
    }

    const float bsv = bias[n0 + m];
#pragma unroll
    for (int t = 0; t < MT; ++t) {
#pragma unroll
        for (int r = 0; r < 8; ++r) {
            float v = acc[t][r] + bsv;
            v = v > 0.f ? v : 0.f;   // ReLU
            ob[t][(size_t)r * CO] = f2bf(v);
        }
    }
}

// ---------------------------------------------------------------------------
// 2x2 max pool, NHWC bf16 (inputs already ReLU'd)
// ---------------------------------------------------------------------------
__global__ void maxpool2_nhwc(const unsigned short* __restrict__ in,
                              unsigned short* __restrict__ out,
                              int C, int Hout, int Wout) {
    int i = blockIdx.x * 256 + threadIdx.x;
    int total = 8 * Hout * Wout * C;
    if (i >= total) return;
    int c = i % C;   int pos = i / C;
    int x = pos % Wout; int tmp = pos / Wout;
    int y = tmp % Hout; int b = tmp / Hout;
    int Win = Wout * 2;
    const unsigned short* r0 =
        in + (((size_t)(b * Hout * 2) + 2 * y) * Win + 2 * x) * C + c;
    size_t rowstep = (size_t)Win * C;
    float v = fmaxf(fmaxf(bf2f(r0[0]), bf2f(r0[C])),
                    fmaxf(bf2f(r0[rowstep]), bf2f(r0[rowstep + C])));
    out[i] = f2bf(v);
}

// ---------------------------------------------------------------------------
// Persistent decoder: one 8-wave workgroup runs all 49 GRU+attention steps.
// LDS activations are bf16 with 9 rows: rows 0..7 live data, row 8 is a
// shared permanently-zero row used by lanes 8..15 of each A fragment.
// ---------------------------------------------------------------------------
struct DecParams {
    const unsigned short* feat;   // (8,16,32,128) bf16 NHWC
    float* out;                   // (8,50,128) f32
    float* from_a;                // (8,512) ws
    float* alpha;                 // (8,512) ws
    float* beta;                  // (8,512) ws
    const unsigned short *embed_w, *Wyz_w, *Uhz_w, *Ccz_w, *Wyr_w, *Uhr_w,
                         *Ccr_w, *Wyh_w, *Urh_w, *Wo_w, *Wh_w, *Wc_w;
    const float *embed_b, *Wyz_b, *Uhz_b, *Ccz_b, *Wyr_b, *Uhr_b, *Ccr_b,
                *Wyh_b, *Urh_b, *Wo_b, *Wh_b, *Wc_b,
                *cov_h_w, *cov_h_b, *cov_a_w, *cov_a_b, *cov_b_w, *cov_b_b,
                *convq_w, *convq_b;
};

// One 16x16 tile of Out(8 x O) += A(8 x K, bf16 LDS) @ W(O,K bf16)^T
__device__ __forceinline__ v8f gemm_tile_acc_bf(const unsigned short* __restrict__ A,
                                                int K,
                                                const unsigned short* __restrict__ W,
                                                int n0, int lane, v8f acc) {
    const int m = lane & 15;
    const int g = lane >> 4;
    const unsigned short* arow = A + (m < 8 ? m : 8) * K + g * 16;
    const unsigned short* wrow = W + (size_t)(n0 + m) * K + g * 16;
    for (int kb = 0; kb < K; kb += 32) {
        BF16x16 af, bf;
        af.q[0] = *(const uint4*)(arow + kb);
        af.q[1] = *(const uint4*)(arow + kb + 8);
        bf.q[0] = *(const uint4*)(wrow + kb);
        bf.q[1] = *(const uint4*)(wrow + kb + 8);
        acc = __builtin_amdgcn_wmma_f32_16x16x32_bf16(
                  false, af.v, false, bf.v, (short)0, acc, false, false);
    }
    return acc;
}

// Dual-output GEMM: two weight matrices share every A fragment (z & r gates).
__device__ __forceinline__ void gemm_tile_acc2_bf(const unsigned short* __restrict__ A,
                                                  int K,
                                                  const unsigned short* __restrict__ W1,
                                                  const unsigned short* __restrict__ W2,
                                                  int n0, int lane,
                                                  v8f& a1, v8f& a2) {
    const int m = lane & 15;
    const int g = lane >> 4;
    const unsigned short* arow = A + (m < 8 ? m : 8) * K + g * 16;
    const unsigned short* w1 = W1 + (size_t)(n0 + m) * K + g * 16;
    const unsigned short* w2 = W2 + (size_t)(n0 + m) * K + g * 16;
    for (int kb = 0; kb < K; kb += 32) {
        BF16x16 af, b1, b2;
        af.q[0] = *(const uint4*)(arow + kb);
        af.q[1] = *(const uint4*)(arow + kb + 8);
        b1.q[0] = *(const uint4*)(w1 + kb);
        b1.q[1] = *(const uint4*)(w1 + kb + 8);
        b2.q[0] = *(const uint4*)(w2 + kb);
        b2.q[1] = *(const uint4*)(w2 + kb + 8);
        a1 = __builtin_amdgcn_wmma_f32_16x16x32_bf16(
                 false, af.v, false, b1.v, (short)0, a1, false, false);
        a2 = __builtin_amdgcn_wmma_f32_16x16x32_bf16(
                 false, af.v, false, b2.v, (short)0, a2, false, false);
    }
}

// Store tile rows 0..7 into bf16 LDS; bias from pre-summed LDS vector.
// act: 0=none, 1=sigmoid, 2=tanh (compile-time at call sites).
__device__ __forceinline__ void store_act_bf(unsigned short* O, int Od, int n0,
                                             int lane, v8f acc,
                                             const float* biasvec, int act) {
    const int m = lane & 15, g = lane >> 4;
    if (g == 0) {
        const float bv = biasvec[n0 + m];
#pragma unroll
        for (int r = 0; r < 8; ++r) {
            float v = acc[r] + bv;
            if (act == 1) v = 1.f / (1.f + expf(-v));
            else if (act == 2) v = tanhf(v);
            O[r * Od + n0 + m] = f2bf(v);
        }
    }
}

__global__ __launch_bounds__(256) void decoder_kernel(DecParams P) {
    __shared__ alignas(16) unsigned short sh_h[9 * 512];
    __shared__ alignas(16) unsigned short sA[9 * 512];   // z gate
    __shared__ alignas(16) unsigned short sB[9 * 512];   // r -> r*h -> h_cand
    __shared__ alignas(16) unsigned short sE[9 * 128];
    __shared__ alignas(16) unsigned short sCtx[9 * 128];
    __shared__ alignas(16) unsigned short sT[9 * 128];
    __shared__ float sBz[512], sBr[512], sBh[512];       // pre-summed biases
    __shared__ float sBe[128], sBt[128], sBo[128];
    __shared__ float sKeff[10];
    __shared__ float sFh[8];

    const int tid  = threadIdx.x;
    const int lane = tid & 31;
    const int wave = tid >> 5;

    // ---- one-time init -------------------------------------------------
    for (int i = tid; i < 9 * 512; i += 256) { sh_h[i] = 0; sA[i] = 0; sB[i] = 0; }
    for (int i = tid; i < 9 * 128; i += 256) { sE[i] = 0; sCtx[i] = 0; sT[i] = 0; }
    for (int i = tid; i < 512; i += 256) {
        sBz[i] = P.Wyz_b[i] + P.Uhz_b[i] + P.Ccz_b[i];
        sBr[i] = P.Wyr_b[i] + P.Uhr_b[i] + P.Ccr_b[i];
        sBh[i] = P.Wyh_b[i] + P.Urh_b[i] + P.Ccz_b[i];
    }
    for (int i = tid; i < 128; i += 256) {
        sBe[i] = P.embed_b[i];
        sBt[i] = P.Wh_b[i] + P.Wc_b[i];
        sBo[i] = P.Wo_b[i];
    }
    for (int i = tid; i < 4096; i += 256) {
        P.alpha[i] = 1.0f / 512.0f;
        P.beta[i]  = 0.f;
    }
    for (int i = tid; i < 1024; i += 256) {
        int b = i >> 7, o = i & 127;
        P.out[(b * 50) * 128 + o] = (o == 0) ? 1.f : 0.f;   // one-hot <s>
    }
    for (int i = tid; i < 4096; i += 256) {                  // from_a
        int b = i >> 9, p = i & 511;
        const unsigned short* fr = P.feat + ((size_t)(b * 512) + p) * 128;
        float s = P.cov_a_b[0];
        for (int c = 0; c < 128; c += 2) {
            unsigned int qq = *(const unsigned int*)(fr + c);
            s += __uint_as_float(qq << 16) * P.cov_a_w[c];
            s += __uint_as_float(qq & 0xffff0000u) * P.cov_a_w[c + 1];
        }
        P.from_a[i] = s;
    }
    if (tid < 9) {                                           // folded K_eff taps
        float s = 0.f;
        for (int q = 0; q < 128; ++q) s += P.cov_b_w[q] * P.convq_w[q * 9 + tid];
        sKeff[tid] = s;
    }
    if (tid == 9) {
        float s = P.cov_b_b[0];
        for (int q = 0; q < 128; ++q) s += P.cov_b_w[q] * P.convq_b[q];
        sKeff[9] = s;
    }
    __threadfence_block();
    __syncthreads();

    // ---- recurrence ----------------------------------------------------
    for (int t = 1; t < 50; ++t) {
        // P0: y_prev -> sT (bf16)
        for (int i = tid; i < 1024; i += 256) {
            int b = i >> 7, o = i & 127;
            sT[i] = f2bf(P.out[((b * 50) + (t - 1)) * 128 + o]);
        }
        // ctx = einsum(feat, alpha): wave b, each lane owns 4 channels
        {
            int b = wave;
            const unsigned short* fb_ =
                P.feat + (size_t)(b * 512) * 128 + lane * 4;
            const float* al = P.alpha + b * 512;
            float s0 = 0.f, s1 = 0.f, s2 = 0.f, s3 = 0.f;
            for (int p = 0; p < 512; ++p) {
                uint2 qq = *(const uint2*)(fb_ + (size_t)p * 128);
                float a = al[p];
                s0 += __uint_as_float(qq.x << 16) * a;
                s1 += __uint_as_float(qq.x & 0xffff0000u) * a;
                s2 += __uint_as_float(qq.y << 16) * a;
                s3 += __uint_as_float(qq.y & 0xffff0000u) * a;
            }
            sCtx[b * 128 + lane * 4 + 0] = f2bf(s0);
            sCtx[b * 128 + lane * 4 + 1] = f2bf(s1);
            sCtx[b * 128 + lane * 4 + 2] = f2bf(s2);
            sCtx[b * 128 + lane * 4 + 3] = f2bf(s3);
        }
        __syncthreads();

        // P1: e = y @ embed^T + b
        {
            int n0 = wave * 16;
            v8f a = zero8();
            a = gemm_tile_acc_bf(sT, 128, P.embed_w, n0, lane, a);
            store_act_bf(sE, 128, n0, lane, a, sBe, 0);
        }
        __syncthreads();

        // P2: z and r gates share A fragments (dual GEMMs), 4 tiles per wave
#pragma unroll
        for (int j = 0; j < 4; ++j) {
            int n0 = (wave + 8 * j) * 16;
            v8f az = zero8(), ar = zero8();
            gemm_tile_acc2_bf(sE,   128, P.Wyz_w, P.Wyr_w, n0, lane, az, ar);
            gemm_tile_acc2_bf(sh_h, 512, P.Uhz_w, P.Uhr_w, n0, lane, az, ar);
            gemm_tile_acc2_bf(sCtx, 128, P.Ccz_w, P.Ccr_w, n0, lane, az, ar);
            store_act_bf(sA, 512, n0, lane, az, sBz, 1);
            store_act_bf(sB, 512, n0, lane, ar, sBr, 1);
        }
        __syncthreads();

        // P3: sB = r * h
        for (int i = tid; i < 4096; i += 256)
            sB[i] = f2bf(bf2f(sB[i]) * bf2f(sh_h[i]));
        __syncthreads();

        // P4: h_cand = tanh(e@Wyh^T + (r*h)@Urh^T + ctx@Ccz^T)  [shares Ccz]
        {
            v8f ah[4];
#pragma unroll
            for (int j = 0; j < 4; ++j) {
                int n0 = (wave + 8 * j) * 16;
                v8f a = zero8();
                a = gemm_tile_acc_bf(sE,   128, P.Wyh_w, n0, lane, a);
                a = gemm_tile_acc_bf(sB,   512, P.Urh_w, n0, lane, a);
                a = gemm_tile_acc_bf(sCtx, 128, P.Ccz_w, n0, lane, a);
                ah[j] = a;
            }
            __syncthreads();   // finish reading rh before overwriting sB
#pragma unroll
            for (int j = 0; j < 4; ++j) {
                int n0 = (wave + 8 * j) * 16;
                store_act_bf(sB, 512, n0, lane, ah[j], sBh, 2);
            }
        }
        __syncthreads();

        // P5: h = (1-z)*h + z*hc
        for (int i = tid; i < 4096; i += 256) {
            float z = bf2f(sA[i]);
            sh_h[i] = f2bf((1.f - z) * bf2f(sh_h[i]) + z * bf2f(sB[i]));
        }
        __syncthreads();

        // P6: sT = e + h@Wh^T + ctx@Wc^T + biases
        {
            int n0 = wave * 16;
            v8f a = zero8();
            a = gemm_tile_acc_bf(sh_h, 512, P.Wh_w, n0, lane, a);
            a = gemm_tile_acc_bf(sCtx, 128, P.Wc_w, n0, lane, a);
            const int m = lane & 15, g = lane >> 4;
            if (g == 0) {
                const float bv = sBt[n0 + m];
#pragma unroll
                for (int r = 0; r < 8; ++r)
                    sT[r * 128 + n0 + m] =
                        f2bf(a[r] + bf2f(sE[r * 128 + n0 + m]) + bv);
            }
        }
        __syncthreads();

        // P7: y = sT @ Wo^T + Wo_b  -> out row t (f32 global)
        {
            int n0 = wave * 16;
            v8f a = zero8();
            a = gemm_tile_acc_bf(sT, 128, P.Wo_w, n0, lane, a);
            const int m = lane & 15, g = lane >> 4;
            if (g == 0) {
                const float bv = sBo[n0 + m];
#pragma unroll
                for (int r = 0; r < 8; ++r)
                    P.out[((r * 50) + t) * 128 + n0 + m] = a[r] + bv;
            }
        }

        // P8: beta += alpha ; from_h = h @ cov_h^T + b
        for (int i = tid; i < 4096; i += 256) P.beta[i] += P.alpha[i];
        {
            int b = wave;
            float s = 0.f;
            for (int k2 = lane; k2 < 512; k2 += 32)
                s += bf2f(sh_h[b * 512 + k2]) * P.cov_h_w[k2];
            for (int off = 16; off > 0; off >>= 1) s += __shfl_xor(s, off, 32);
            if (lane == 0) sFh[b] = s + P.cov_h_b[0];
        }
        __threadfence_block();
        __syncthreads();

        // P9: att = tanh(from_a + Keff*beta + from_h); alpha = softmax(att)
        {
            int b = wave;
            float att_l[16];
            float mx = -1e30f;
            for (int i2 = 0; i2 < 16; ++i2) {
                int p  = lane + 32 * i2;
                int hh = p >> 5, ww = p & 31;
                float fb = sKeff[9];
#pragma unroll
                for (int ky = 0; ky < 3; ++ky) {
                    int iy = hh + ky - 1;
                    if (iy < 0 || iy >= 16) continue;
#pragma unroll
                    for (int kx = 0; kx < 3; ++kx) {
                        int ix = ww + kx - 1;
                        if (ix < 0 || ix >= 32) continue;
                        fb += sKeff[ky * 3 + kx] * P.beta[b * 512 + iy * 32 + ix];
                    }
                }
                float a = tanhf(P.from_a[b * 512 + p] + fb + sFh[b]);
                att_l[i2] = a;
                mx = fmaxf(mx, a);
            }
            for (int off = 16; off > 0; off >>= 1)
                mx = fmaxf(mx, __shfl_xor(mx, off, 32));
            float ex[16], ssum = 0.f;
            for (int i2 = 0; i2 < 16; ++i2) {
                ex[i2] = expf(att_l[i2] - mx);
                ssum += ex[i2];
            }
            for (int off = 16; off > 0; off >>= 1) ssum += __shfl_xor(ssum, off, 32);
            float inv = 1.f / ssum;
            for (int i2 = 0; i2 < 16; ++i2)
                P.alpha[b * 512 + lane + 32 * i2] = ex[i2] * inv;
        }
        __threadfence_block();
        __syncthreads();
    }
}

// ---------------------------------------------------------------------------
// Host orchestration
// ---------------------------------------------------------------------------
extern "C" void kernel_launch(void* const* d_in, const int* in_sizes, int n_in,
                              void* d_out, int out_size, void* d_ws, size_t ws_size,
                              hipStream_t stream) {
    (void)in_sizes; (void)n_in; (void)out_size; (void)ws_size;

    const float* x = (const float*)d_in[0];
    const float* cw[16]; const float* cb[16];
    for (int i = 0; i < 16; ++i) {
        cw[i] = (const float*)d_in[1 + 2 * i];
        cb[i] = (const float*)d_in[2 + 2 * i];
    }

    // workspace carve
    const size_t ACT = (size_t)8 * 256 * 512 * 32;   // max NHWC elems per tensor
    char* ws = (char*)d_ws;
    unsigned short* bufA = (unsigned short*)ws;
    unsigned short* bufB = (unsigned short*)(ws + ACT * 2);
    float* from_a = (float*)(ws + ACT * 4);
    float* alpha  = from_a + 8 * 512;
    float* beta   = alpha + 8 * 512;
    unsigned short* cur = (unsigned short*)(ws + ACT * 4 + 48 * 1024);

    // ---- pack weights (bf16) -------------------------------------------
    const int CIs[16]  = {9,32,32,32, 32,64,64,64, 64,64,64,64, 64,128,128,128};
    const int CIPs[16] = {32,32,32,32, 32,64,64,64, 64,64,64,64, 64,128,128,128};
    const int COs[16]  = {32,32,32,32, 64,64,64,64, 64,64,64,64, 128,128,128,128};
    unsigned short* wp[16];
    for (int i = 0; i < 16; ++i) {
        int n = 9 * COs[i] * CIPs[i];
        wp[i] = cur;
        pack_conv_w<<<dim3((n + 255) / 256), dim3(256), 0, stream>>>(
            cw[i], cur, COs[i], CIs[i], CIPs[i]);
        cur += n;
    }
    const int lin_idx[12]   = {33, 35, 37, 39, 41, 43, 45, 47, 49, 51, 53, 55};
    const int lin_elems[12] = {16384, 65536, 262144, 65536, 65536, 262144,
                               65536, 65536, 262144, 16384, 65536, 16384};
    unsigned short* linp[12];
    for (int i = 0; i < 12; ++i) {
        linp[i] = cur;
        f32_to_bf16_kernel<<<dim3((lin_elems[i] + 255) / 256), dim3(256), 0, stream>>>(
            (const float*)d_in[lin_idx[i]], cur, lin_elems[i]);
        cur += lin_elems[i];
    }
    unsigned short* zbuf = cur;                       // 256-elem zero line
    zero_u16_kernel<<<dim3(1), dim3(256), 0, stream>>>(zbuf, 256);
    cur += 256;

    // ---- encoder --------------------------------------------------------
    {
        int n = 8 * 256 * 512 * 32;
        pack_input_nhwc<<<dim3((n + 255) / 256), dim3(256), 0, stream>>>(
            x, bufA, 256, 512);
    }
    dim3 wv(32);
    // block 1 @ 256x512 (input padded 9 -> 32 ch)
    conv3x3_nhwc_wmma<32, 32, 4, 256, 512><<<dim3(16384, 2), wv, 0, stream>>>(bufA, wp[0], cb[0], bufB, zbuf);
    conv3x3_nhwc_wmma<32, 32, 4, 256, 512><<<dim3(16384, 2), wv, 0, stream>>>(bufB, wp[1], cb[1], bufA, zbuf);
    conv3x3_nhwc_wmma<32, 32, 4, 256, 512><<<dim3(16384, 2), wv, 0, stream>>>(bufA, wp[2], cb[2], bufB, zbuf);
    conv3x3_nhwc_wmma<32, 32, 4, 256, 512><<<dim3(16384, 2), wv, 0, stream>>>(bufB, wp[3], cb[3], bufA, zbuf);
    maxpool2_nhwc<<<dim3((8 * 128 * 256 * 32 + 255) / 256), dim3(256), 0, stream>>>(bufA, bufB, 32, 128, 256);
    // block 2 @ 128x256
    conv3x3_nhwc_wmma<32, 64, 4, 128, 256><<<dim3(4096, 4), wv, 0, stream>>>(bufB, wp[4], cb[4], bufA, zbuf);
    conv3x3_nhwc_wmma<64, 64, 4, 128, 256><<<dim3(4096, 4), wv, 0, stream>>>(bufA, wp[5], cb[5], bufB, zbuf);
    conv3x3_nhwc_wmma<64, 64, 4, 128, 256><<<dim3(4096, 4), wv, 0, stream>>>(bufB, wp[6], cb[6], bufA, zbuf);
    conv3x3_nhwc_wmma<64, 64, 4, 128, 256><<<dim3(4096, 4), wv, 0, stream>>>(bufA, wp[7], cb[7], bufB, zbuf);
    maxpool2_nhwc<<<dim3((8 * 64 * 128 * 64 + 255) / 256), dim3(256), 0, stream>>>(bufB, bufA, 64, 64, 128);
    // block 3 @ 64x128
    conv3x3_nhwc_wmma<64, 64, 4, 64, 128><<<dim3(1024, 4), wv, 0, stream>>>(bufA, wp[8], cb[8], bufB, zbuf);
    conv3x3_nhwc_wmma<64, 64, 4, 64, 128><<<dim3(1024, 4), wv, 0, stream>>>(bufB, wp[9], cb[9], bufA, zbuf);
    conv3x3_nhwc_wmma<64, 64, 4, 64, 128><<<dim3(1024, 4), wv, 0, stream>>>(bufA, wp[10], cb[10], bufB, zbuf);
    conv3x3_nhwc_wmma<64, 64, 4, 64, 128><<<dim3(1024, 4), wv, 0, stream>>>(bufB, wp[11], cb[11], bufA, zbuf);
    maxpool2_nhwc<<<dim3((8 * 32 * 64 * 64 + 255) / 256), dim3(256), 0, stream>>>(bufA, bufB, 64, 32, 64);
    // block 4 @ 32x64
    conv3x3_nhwc_wmma<64, 128, 4, 32, 64><<<dim3(256, 8), wv, 0, stream>>>(bufB, wp[12], cb[12], bufA, zbuf);
    conv3x3_nhwc_wmma<128, 128, 4, 32, 64><<<dim3(256, 8), wv, 0, stream>>>(bufA, wp[13], cb[13], bufB, zbuf);
    conv3x3_nhwc_wmma<128, 128, 4, 32, 64><<<dim3(256, 8), wv, 0, stream>>>(bufB, wp[14], cb[14], bufA, zbuf);
    conv3x3_nhwc_wmma<128, 128, 4, 32, 64><<<dim3(256, 8), wv, 0, stream>>>(bufA, wp[15], cb[15], bufB, zbuf);
    maxpool2_nhwc<<<dim3((8 * 16 * 32 * 128 + 255) / 256), dim3(256), 0, stream>>>(bufB, bufA, 128, 16, 32);
    // feat = bufA (8,16,32,128) NHWC bf16

    // ---- decoder --------------------------------------------------------
    DecParams P;
    P.feat = bufA; P.out = (float*)d_out;
    P.from_a = from_a; P.alpha = alpha; P.beta = beta;
    P.embed_w = linp[0]; P.Wyz_w = linp[1]; P.Uhz_w = linp[2]; P.Ccz_w = linp[3];
    P.Wyr_w = linp[4];   P.Uhr_w = linp[5]; P.Ccr_w = linp[6]; P.Wyh_w = linp[7];
    P.Urh_w = linp[8];   P.Wo_w  = linp[9]; P.Wh_w  = linp[10]; P.Wc_w = linp[11];
    P.embed_b = (const float*)d_in[34];
    P.Wyz_b = (const float*)d_in[36];
    P.Uhz_b = (const float*)d_in[38];
    P.Ccz_b = (const float*)d_in[40];
    P.Wyr_b = (const float*)d_in[42];
    P.Uhr_b = (const float*)d_in[44];
    P.Ccr_b = (const float*)d_in[46];
    P.Wyh_b = (const float*)d_in[48];
    P.Urh_b = (const float*)d_in[50];
    P.Wo_b  = (const float*)d_in[52];
    P.Wh_b  = (const float*)d_in[54];
    P.Wc_b  = (const float*)d_in[56];
    P.cov_h_w = (const float*)d_in[57]; P.cov_h_b = (const float*)d_in[58];
    P.cov_a_w = (const float*)d_in[59]; P.cov_a_b = (const float*)d_in[60];
    P.cov_b_w = (const float*)d_in[61]; P.cov_b_b = (const float*)d_in[62];
    P.convq_w = (const float*)d_in[63]; P.convq_b = (const float*)d_in[64];

    decoder_kernel<<<dim3(1), dim3(256), 0, stream>>>(P);
}
